// MoELayer_52201032515790
// MI455X (gfx1250) — matrix operand order repaired and checked by reading, not compiled, not run on hardware
//
#include <hip/hip_runtime.h>
#include <hip/hip_bf16.h>

// ---------------------------------------------------------------------------
// MoE top-2 layer for MI455X (gfx1250), wave32 + V_WMMA_F32_16X16X32_BF16.
// fp32 kept via bf16 hi/lo split: 3 bf16 WMMAs per C tile ~= fp32 accuracy.
// Fragments fed by aligned ds_load_b128 (LDS row stride 80B = 5x16B).
// ---------------------------------------------------------------------------

#define NTOK   8192      // B*S
#define DIM    1024      // D
#define NOUT   1024      // OUT
#define NEXP   8
#define TOPK   2
#define TOTALP (NTOK * TOPK)   // 16384 permuted slots

#define M_TILE 64
#define N_TILE 128
#define K_TILE 32
#define MAX_TILES (TOTALP / M_TILE + NEXP)   // 264

#define ASTR 40   // LDS row stride in bf16 elems: 80B = 5 x 16B, b128-aligned
#define BSTR 40

typedef __attribute__((ext_vector_type(16))) __bf16 v16bf;
typedef __attribute__((ext_vector_type(8)))  float  v8f;

union FragU {
    v16bf v;
    uint4 q[2];
};

static __device__ __forceinline__ unsigned short f32_to_bf16_bits(float f) {
    union { __bf16 h; unsigned short u; } cv;
    cv.h = (__bf16)f;              // RNE convert
    return cv.u;
}
static __device__ __forceinline__ float bf16_bits_to_f32(unsigned short b) {
    union { __bf16 h; unsigned short u; } cv;
    cv.u = b;
    return (float)cv.h;
}
// split one f32 into (hi, lo) bf16 bit patterns
static __device__ __forceinline__ void split_bf16(float v, unsigned short& hb,
                                                  unsigned short& lb) {
    hb = f32_to_bf16_bits(v);
    lb = f32_to_bf16_bits(v - bf16_bits_to_f32(hb));
}

// ---------------------------------------------------------------------------
// 0) zero the small int header of the workspace (counts / offsets / num_tiles)
// ---------------------------------------------------------------------------
__global__ void moe_init(int* wsi) {
    int i = threadIdx.x;
    if (i < 32) wsi[i] = 0;
}

// ---------------------------------------------------------------------------
// 1) router: logits = x @ Wg + bg, softmax over E=8, top-2, renormalize.
//    one wave32 per token; lanes split D, __shfl_xor tree reduction.
// ---------------------------------------------------------------------------
__global__ __launch_bounds__(256) void moe_router(
    const float* __restrict__ x, const float* __restrict__ Wg,
    const float* __restrict__ bg,
    int* __restrict__ topk_idx, int* __restrict__ topk_rank,
    float* __restrict__ topk_gate, int* __restrict__ counts) {

    const int wave = threadIdx.x >> 5;
    const int lane = threadIdx.x & 31;
    const int t    = blockIdx.x * 8 + wave;     // token id, grid = 1024 blocks
    const float* xr = x + (size_t)t * DIM;

    float acc[NEXP];
#pragma unroll
    for (int e = 0; e < NEXP; ++e) acc[e] = 0.f;

    for (int d = lane; d < DIM; d += 32) {
        float xv = xr[d];
        const float4* w = (const float4*)(Wg + (size_t)d * NEXP);
        float4 w0 = w[0], w1 = w[1];
        acc[0] += xv * w0.x; acc[1] += xv * w0.y;
        acc[2] += xv * w0.z; acc[3] += xv * w0.w;
        acc[4] += xv * w1.x; acc[5] += xv * w1.y;
        acc[6] += xv * w1.z; acc[7] += xv * w1.w;
    }
#pragma unroll
    for (int off = 16; off >= 1; off >>= 1) {
#pragma unroll
        for (int e = 0; e < NEXP; ++e)
            acc[e] += __shfl_xor(acc[e], off, 32);
    }

    if (lane == 0) {
#pragma unroll
        for (int e = 0; e < NEXP; ++e) acc[e] += bg[e];
        // top-1 / top-2 (ties -> lowest index, matching lax.top_k)
        int i1 = 0;
#pragma unroll
        for (int e = 1; e < NEXP; ++e) if (acc[e] > acc[i1]) i1 = e;
        int i2 = (i1 == 0) ? 1 : 0;
#pragma unroll
        for (int e = 0; e < NEXP; ++e)
            if (e != i1 && acc[e] > acc[i2]) i2 = e;
        // softmax over all 8, then renormalize the top-2 probs
        float mx = acc[i1];
        float den = 0.f;
#pragma unroll
        for (int e = 0; e < NEXP; ++e) den += __expf(acc[e] - mx);
        float p1 = __expf(acc[i1] - mx) / den;
        float p2 = __expf(acc[i2] - mx) / den;
        float s  = p1 + p2 + 1e-9f;
        float g1 = p1 / s, g2 = p2 / s;

        topk_idx[t * 2 + 0]  = i1;
        topk_idx[t * 2 + 1]  = i2;
        topk_gate[t * 2 + 0] = g1;
        topk_gate[t * 2 + 1] = g2;
        topk_rank[t * 2 + 0] = atomicAdd(&counts[i1], 1);
        topk_rank[t * 2 + 1] = atomicAdd(&counts[i2], 1);
    }
}

// ---------------------------------------------------------------------------
// 2) plan: exclusive-scan counts -> offsets; emit (expert, row0) M-tile list.
// ---------------------------------------------------------------------------
__global__ void moe_plan(int* wsi) {
    if (threadIdx.x == 0 && blockIdx.x == 0) {
        int off = 0, nt = 0;
        for (int e = 0; e < NEXP; ++e) {
            wsi[8 + e] = off;                 // offsets
            int c = wsi[e];                   // counts
            for (int r = 0; r < c; r += M_TILE)
                wsi[32 + nt++] = (e << 16) | r;
            off += c;
        }
        wsi[16] = nt;                         // num_tiles (<= MAX_TILES)
    }
}

// ---------------------------------------------------------------------------
// 3) scatter: slot = offsets[e] + rank; build gather list + per-token slots.
// ---------------------------------------------------------------------------
__global__ __launch_bounds__(256) void moe_scatter(
    const int* __restrict__ topk_idx, const int* __restrict__ topk_rank,
    const int* __restrict__ offsets,
    int* __restrict__ perm_token, int* __restrict__ slot_of) {
    int g = blockIdx.x * 256 + threadIdx.x;   // 0 .. TOTALP-1
    if (g < TOTALP) {
        int t = g >> 1;
        int e = topk_idx[g];
        int slot = offsets[e] + topk_rank[g];
        perm_token[slot] = t;
        slot_of[g] = slot;
    }
}

// ---------------------------------------------------------------------------
// 4) grouped GEMM: for each (expert M-tile, N-tile) compute
//       y_perm[slot, n] = x[tok] @ We[e] + be[e]
//    64x128 block tile, 8 waves -> each wave owns 16x64 (4 WMMA C tiles).
//    fp32 -> bf16 hi/lo split in LDS; 3x v_wmma_f32_16x16x32_bf16 per tile.
//    All fragment loads are 16B-aligned ds_load_b128.
// ---------------------------------------------------------------------------
__global__ __launch_bounds__(256) void moe_gemm(
    const float* __restrict__ x, const float* __restrict__ We,
    const float* __restrict__ be,
    const int* __restrict__ perm_token,
    const int* __restrict__ num_tiles, const int* __restrict__ tiles,
    const int* __restrict__ offsets, const int* __restrict__ counts,
    float* __restrict__ y) {

    const int bx = blockIdx.x;
    if (bx >= *num_tiles) return;             // uniform per block

    const int info      = tiles[bx];
    const int e         = info >> 16;
    const int row_local = info & 0xFFFF;
    const int cnt       = counts[e];
    const int base      = offsets[e];
    int rows_valid      = cnt - row_local;
    if (rows_valid > M_TILE) rows_valid = M_TILE;
    const int n0 = blockIdx.y * N_TILE;

    const int tid  = threadIdx.x;
    const int wave = tid >> 5;
    const int lane = tid & 31;
    const int wm   = wave & 3;                // 4 M sub-blocks of 16 rows
    const int wn   = wave >> 2;               // 2 N sub-blocks of 64 cols
    const int m    = lane & 15;
    const int h    = lane >> 4;

    __shared__ __align__(16) unsigned short Ahi[M_TILE * ASTR];
    __shared__ __align__(16) unsigned short Alo[M_TILE * ASTR];
    __shared__ __align__(16) unsigned short Bhi[N_TILE * BSTR];
    __shared__ __align__(16) unsigned short Blo[N_TILE * BSTR];
    __shared__ int toks[M_TILE];

    // preload gathered token ids once per block
    if (tid < M_TILE)
        toks[tid] = (tid < rows_valid) ? perm_token[base + row_local + tid] : -1;

    const float* Wbase = We + (size_t)e * DIM * NOUT;

    v8f c[4] = {};                            // 4 x 16x16 f32 accumulators

    for (int k0 = 0; k0 < DIM; k0 += K_TILE) {
        __syncthreads();
        // ---- stage A tile: 64 rows x 32 K; float2 loads, packed b32 stores
#pragma unroll
        for (int t = 0; t < 4; ++t) {
            int idx = tid + t * 256;          // 0..1023 (pairs)
            int r   = idx >> 4;               // row 0..63
            int kp  = (idx & 15) * 2;         // k pair start 0..30
            float v0 = 0.f, v1 = 0.f;
            int tok = toks[r];
            if (tok >= 0) {
                const float2 xv =
                    *(const float2*)&x[(size_t)tok * DIM + k0 + kp];
                v0 = xv.x; v1 = xv.y;
            }
            unsigned short h0, l0, h1, l1;
            split_bf16(v0, h0, l0);
            split_bf16(v1, h1, l1);
            *(unsigned int*)&Ahi[r * ASTR + kp] = (unsigned)h0 | ((unsigned)h1 << 16);
            *(unsigned int*)&Alo[r * ASTR + kp] = (unsigned)l0 | ((unsigned)l1 << 16);
        }
        // ---- stage B tile: 32 K x 128 N, transposed (K-contiguous rows);
        //      read vertical d-pairs so each store is a packed b32
#pragma unroll
        for (int t = 0; t < 8; ++t) {
            int idx = tid + t * 256;          // 0..2047 (pairs)
            int dl  = (idx >> 7) * 2;         // d pair start 0..30
            int fl  = idx & 127;              // out column 0..127
            float v0 = Wbase[(size_t)(k0 + dl) * NOUT + n0 + fl];
            float v1 = Wbase[(size_t)(k0 + dl + 1) * NOUT + n0 + fl];
            unsigned short h0, l0, h1, l1;
            split_bf16(v0, h0, l0);
            split_bf16(v1, h1, l1);
            *(unsigned int*)&Bhi[fl * BSTR + dl] = (unsigned)h0 | ((unsigned)h1 << 16);
            *(unsigned int*)&Blo[fl * BSTR + dl] = (unsigned)l0 | ((unsigned)l1 << 16);
        }
        __syncthreads();

        // ---- A fragment (ISA 16-bit 16x32 layout): two b128 loads each
        FragU a_hi, a_lo;
        {
            const int rb = (wm * 16 + m) * ASTR;
            a_hi.q[0] = *(const uint4*)&Ahi[rb + h * 8];        // K g=0 chunk
            a_hi.q[1] = *(const uint4*)&Ahi[rb + 16 + h * 8];   // K g=1 chunk
            a_lo.q[0] = *(const uint4*)&Alo[rb + h * 8];
            a_lo.q[1] = *(const uint4*)&Alo[rb + 16 + h * 8];
        }
        // ---- 4 N sub-tiles of 16: B frag = two b128 loads (same K layout as A)
#pragma unroll
        for (int s = 0; s < 4; ++s) {
            const int nb = (wn * 64 + s * 16 + m) * BSTR + h * 8;
            FragU b_hi, b_lo;
            b_hi.q[0] = *(const uint4*)&Bhi[nb];
            b_hi.q[1] = *(const uint4*)&Bhi[nb + 16];
            b_lo.q[0] = *(const uint4*)&Blo[nb];
            b_lo.q[1] = *(const uint4*)&Blo[nb + 16];
            // fp32 ~= hi*hi + hi*lo + lo*hi  (lo*lo below rounding)
            c[s] = __builtin_amdgcn_wmma_f32_16x16x32_bf16(
                false, a_hi.v, false, b_hi.v, (short)0, c[s], false, false);
            c[s] = __builtin_amdgcn_wmma_f32_16x16x32_bf16(
                false, a_hi.v, false, b_lo.v, (short)0, c[s], false, false);
            c[s] = __builtin_amdgcn_wmma_f32_16x16x32_bf16(
                false, a_lo.v, false, b_hi.v, (short)0, c[s], false, false);
        }
    }

    // ---- epilogue: + bias, store to permuted slot buffer
#pragma unroll
    for (int s = 0; s < 4; ++s) {
#pragma unroll
        for (int v = 0; v < 8; ++v) {
            int mm = wm * 16 + v + 8 * h;     // C layout: VGPR v -> M = v + 8h
            if (mm < rows_valid) {
                int ncol = n0 + wn * 64 + s * 16 + m;
                y[(size_t)(base + row_local + mm) * NOUT + ncol] =
                    c[s][v] + be[e * NOUT + ncol];
            }
        }
    }
}

// ---------------------------------------------------------------------------
// 5) combine: out[t] = g0*y[slot0] + g1*y[slot1]   (gates multiply bias too)
// ---------------------------------------------------------------------------
__global__ __launch_bounds__(256) void moe_combine(
    const float* __restrict__ y, const int* __restrict__ slot_of,
    const float* __restrict__ gates, float* __restrict__ out) {
    const int t  = blockIdx.x;
    const int i  = threadIdx.x;               // 256 threads x float4 = 1024
    const int s0 = slot_of[t * 2 + 0];
    const int s1 = slot_of[t * 2 + 1];
    const float g0 = gates[t * 2 + 0];
    const float g1 = gates[t * 2 + 1];
    const float4* y0 = (const float4*)(y + (size_t)s0 * NOUT);
    const float4* y1 = (const float4*)(y + (size_t)s1 * NOUT);
    float4* o = (float4*)(out + (size_t)t * NOUT);
    float4 a = y0[i], b = y1[i], r;
    r.x = g0 * a.x + g1 * b.x;
    r.y = g0 * a.y + g1 * b.y;
    r.z = g0 * a.z + g1 * b.z;
    r.w = g0 * a.w + g1 * b.w;
    o[i] = r;
}

// ---------------------------------------------------------------------------
// launch
// ---------------------------------------------------------------------------
extern "C" void kernel_launch(void* const* d_in, const int* in_sizes, int n_in,
                              void* d_out, int out_size, void* d_ws, size_t ws_size,
                              hipStream_t stream) {
    const float* x  = (const float*)d_in[0];
    const float* Wg = (const float*)d_in[1];
    const float* bg = (const float*)d_in[2];
    const float* We = (const float*)d_in[3];
    const float* be = (const float*)d_in[4];
    float* out = (float*)d_out;

    // workspace layout (ints unless noted):
    //  [0..7] counts | [8..15] offsets | [16] num_tiles | [32..295] tiles
    //  [512)   topk_idx  (16384)
    //  [16896) topk_rank (16384)
    //  [33280) slot_of   (16384)
    //  [49664) perm_token(16384)
    //  [66048) topk_gate (16384 floats)
    //  [131072) y_perm   (16384*1024 floats = 64 MiB)
    int*   wsi        = (int*)d_ws;
    int*   counts     = wsi;
    int*   offsets    = wsi + 8;
    int*   num_tiles  = wsi + 16;
    int*   tiles      = wsi + 32;
    int*   topk_idx   = wsi + 512;
    int*   topk_rank  = wsi + 16896;
    int*   slot_of    = wsi + 33280;
    int*   perm_token = wsi + 49664;
    float* topk_gate  = (float*)(wsi + 66048);
    float* y_perm     = (float*)d_ws + 131072;

    moe_init<<<1, 64, 0, stream>>>(wsi);
    moe_router<<<NTOK / 8, 256, 0, stream>>>(x, Wg, bg, topk_idx, topk_rank,
                                             topk_gate, counts);
    moe_plan<<<1, 1, 0, stream>>>(wsi);
    moe_scatter<<<TOTALP / 256, 256, 0, stream>>>(topk_idx, topk_rank, offsets,
                                                  perm_token, slot_of);
    dim3 gg(MAX_TILES, NOUT / N_TILE);        // 264 x 8; empty tiles exit fast
    moe_gemm<<<gg, 256, 0, stream>>>(x, We, be, perm_token, num_tiles, tiles,
                                     offsets, counts, y_perm);
    moe_combine<<<NTOK, 256, 0, stream>>>(y_perm, slot_of, topk_gate, out);
}
